// Wavenet_72997264163004
// MI455X (gfx1250) — compile-verified, matrix operand order checked by reading, not run on hardware
//
#include <hip/hip_runtime.h>
#include <hip/hip_bf16.h>
#include <cstdint>
#include <cstddef>

// ---------------- model constants ----------------
#define B_SZ     16
#define T_SZ     4096
#define DIM      256
#define HIDDEN   1024
#define SIZE_IN  8
#define SIZE_OUT 128
#define K_IN     32
#define N_LAYERS 20
#define NPOS     (B_SZ * T_SZ)   // 65536 positions
#define EPSN     1e-6f

// LDS row stride for 256 bf16 + pad (264 u16 = 528 B = 132 dwords -> 16 rows hit
// distinct banks for the 16x32 fragment reads)
#define LPAD 8
#define ASTR (DIM + LPAD)

typedef unsigned short u16;
typedef __attribute__((ext_vector_type(16))) __bf16 bf16x16;
typedef __attribute__((ext_vector_type(8)))  float  f32x8;

union Frag {
    bf16x16 v;
    uint4   q[2];
};

__device__ __forceinline__ u16 f2bf(float f) {
    unsigned u = __float_as_uint(f);
    u += 0x7FFFu + ((u >> 16) & 1u);   // round-to-nearest-even
    return (u16)(u >> 16);
}

__device__ __forceinline__ Frag ldfrag_g(const u16* __restrict__ p) {
    Frag f;
    f.q[0] = *reinterpret_cast<const uint4*>(p);
    f.q[1] = *reinterpret_cast<const uint4*>(p + 16);
    return f;
}

__device__ __forceinline__ f32x8 wmma_bf16(const Frag& a, const Frag& b, f32x8 c) {
    return __builtin_amdgcn_wmma_f32_16x16x32_bf16(
        /*neg_a=*/false, a.v, /*neg_b=*/false, b.v,
        /*c_mod=*/(short)0, c, /*reuse_a=*/false, /*reuse_b=*/false);
}

// CDNA5 async copy global->LDS, 16 bytes per lane, ASYNCcnt-tracked (no VGPR data).
// LDS destination = low 32 bits of the generic shared pointer (flat-LDS aperture).
__device__ __forceinline__ void async_ld_b128(void* lds_ptr, const void* gptr) {
    unsigned lds_off = (unsigned)(uintptr_t)lds_ptr;
    asm volatile("global_load_async_to_lds_b128 %0, %1, off"
                 :: "v"(lds_off), "v"((unsigned long long)(uintptr_t)gptr)
                 : "memory");
}
__device__ __forceinline__ void wait_asynccnt0() {
    asm volatile("s_wait_asynccnt 0x0" ::: "memory");
}

// ---------------- weight conversion ----------------
// W_f / W_g: (L, O=256, I=256, K=2) f32  ->  (L, K, O, I) bf16  (N-major, K-contig)
__global__ __launch_bounds__(256)
void cvt_wfg_kernel(const float* __restrict__ src, u16* __restrict__ dst) {
    int id = blockIdx.x * 256 + threadIdx.x;   // total = L*2*256*256
    int i = id & 255;
    int o = (id >> 8) & 255;
    int k = (id >> 16) & 1;
    int l = id >> 17;
    dst[id] = f2bf(src[(((size_t)l * 256 + o) * 256 + i) * 2 + k]);
}

__global__ __launch_bounds__(256)
void cvt_flat_kernel(const float* __restrict__ src, u16* __restrict__ dst, int n) {
    int id = blockIdx.x * 256 + threadIdx.x;
    if (id < n) dst[id] = f2bf(src[id]);
}

// ---------------- input conv: (B,T,8) -> z (B*T, 256) ----------------
__global__ __launch_bounds__(256)
void input_conv_kernel(const float* __restrict__ x, const float* __restrict__ Win,
                       const float* __restrict__ bin, float* __restrict__ z) {
    __shared__ float xs[(K_IN - 1 + 8) * SIZE_IN];   // 39 rows x 8 ch
    const int tid = threadIdx.x;
    const int b   = (blockIdx.x * 8) / T_SZ;
    const int t0  = (blockIdx.x * 8) % T_SZ;

    for (int e = tid; e < (K_IN - 1 + 8) * SIZE_IN; e += 256) {
        int row = e / SIZE_IN, i = e % SIZE_IN;
        int t = t0 - (K_IN - 1) + row;
        xs[e] = (t >= 0) ? x[((size_t)b * T_SZ + t) * SIZE_IN + i] : 0.f;
    }
    __syncthreads();

    const int c = tid;
    float bc = bin[c];
    float acc[8];
#pragma unroll
    for (int p = 0; p < 8; ++p) acc[p] = bc;
    const float* Wc = Win + (size_t)c * SIZE_IN * K_IN;
    for (int k = 0; k < K_IN; ++k) {
#pragma unroll
        for (int i = 0; i < SIZE_IN; ++i) {
            float w = Wc[i * K_IN + k];
#pragma unroll
            for (int p = 0; p < 8; ++p)
                acc[p] = fmaf(w, xs[(k + p) * SIZE_IN + i], acc[p]);
        }
    }
#pragma unroll
    for (int p = 0; p < 8; ++p)
        z[((size_t)b * T_SZ + t0 + p) * DIM + c] = acc[p];
}

// ---------------- RMS norm over channels, f32 -> bf16 ----------------
// one wave32 per position; lane holds 8 contiguous channels
template <bool RELU>
__global__ __launch_bounds__(256)
void rmsnorm_kernel(const float* __restrict__ src, const float* __restrict__ w,
                    u16* __restrict__ dst) {
    const int wave = threadIdx.x >> 5, lane = threadIdx.x & 31;
    const size_t pos = (size_t)blockIdx.x * 8 + wave;
    const float* p = src + pos * DIM + lane * 8;

    float v[8];
    float ss = 0.f;
#pragma unroll
    for (int i = 0; i < 8; ++i) {
        float t = p[i];
        if (RELU) t = fmaxf(t, 0.f);
        v[i] = t;
        ss += t * t;
    }
#pragma unroll
    for (int o = 16; o > 0; o >>= 1) ss += __shfl_xor(ss, o, 32);
    const float inv = rsqrtf(ss * (1.f / DIM) + EPSN);

    unsigned r[4];
#pragma unroll
    for (int i = 0; i < 4; ++i) {
        float a = v[2 * i]     * inv * w[lane * 8 + 2 * i];
        float b = v[2 * i + 1] * inv * w[lane * 8 + 2 * i + 1];
        r[i] = (unsigned)f2bf(a) | ((unsigned)f2bf(b) << 16);
    }
    uint4 q; q.x = r[0]; q.y = r[1]; q.z = r[2]; q.w = r[3];
    *reinterpret_cast<uint4*>(dst + pos * DIM + lane * 8) = q;
}

// ---------------- fused WaveNet layer ----------------
// block = 8 waves, 32 positions; async-stage zn taps in LDS, f/g WMMA GEMM,
// tanh*sigmoid in-register, u -> LDS, r/s WMMA GEMM, residual/skip RMW.
__global__ __launch_bounds__(256)
void layer_kernel(const u16* __restrict__ zn, float* __restrict__ z,
                  float* __restrict__ sum_out,
                  const u16* __restrict__ Wf, const u16* __restrict__ Wg,   // [2][256][256]
                  const u16* __restrict__ Wr, const u16* __restrict__ Ws,   // [256][256]
                  const float* __restrict__ bf, const float* __restrict__ bg,
                  const float* __restrict__ br, const float* __restrict__ bs,
                  int dil, int first) {
    __shared__ u16 A0[32 * ASTR];   // zn at t - d
    __shared__ u16 A1[32 * ASTR];   // zn at t
    __shared__ u16 U [32 * ASTR];   // gated activations

    const int tid  = threadIdx.x;
    const int pos0 = blockIdx.x * 32;
    const int t0   = pos0 & (T_SZ - 1);
    const int lane = tid & 31;

    // stage 0: async-copy both taps straight into LDS (zero left-pad rows).
    // Each wave stages one full 256-bf16 row per step (16 B/lane), so the
    // pad condition is wave-uniform -> EXEC stays all-ones for async ops.
#pragma unroll
    for (int qi = 0; qi < 4; ++qi) {
        int e   = tid + qi * 256;       // 0..1023 -> 32 rows x 32 uint4
        int row = e >> 5;               // uniform within a wave
        int c8  = (e & 31) * 8;
        async_ld_b128(&A1[row * ASTR + c8],
                      zn + (size_t)(pos0 + row) * DIM + c8);
        if (t0 + row - dil >= 0) {
            async_ld_b128(&A0[row * ASTR + c8],
                          zn + (size_t)(pos0 + row - dil) * DIM + c8);
        } else {
            *reinterpret_cast<uint4*>(&A0[row * ASTR + c8]) = make_uint4(0u, 0u, 0u, 0u);
        }
    }
    wait_asynccnt0();
    __syncthreads();

    const int wave  = tid >> 5;
    const int mtile = wave & 1;
    const int nbase = (wave >> 1) * 64;
    const int lrow  = lane & 15, lhalf = lane >> 4;
    const int arow  = (mtile * 16 + lrow) * ASTR;

    const f32x8 vzero = {};
    f32x8 fa[4], ga[4];
#pragma unroll
    for (int j = 0; j < 4; ++j) { fa[j] = vzero; ga[j] = vzero; }

    // stage 1: f/g GEMM  (K = 2 taps x 256); B-fragments hoisted per K-step
#pragma unroll
    for (int tap = 0; tap < 2; ++tap) {
        const u16* As  = tap ? A1 : A0;                 // folds (tap is unrolled const)
        const u16* Wft = Wf + (size_t)tap * DIM * DIM;
        const u16* Wgt = Wg + (size_t)tap * DIM * DIM;
        for (int k0 = 0; k0 < DIM; k0 += 32) {
            const int ka = k0 + lhalf * 8;
            Frag a;
            a.q[0] = *reinterpret_cast<const uint4*>(&As[arow + ka]);
            a.q[1] = *reinterpret_cast<const uint4*>(&As[arow + ka + 16]);
            Frag bF[4], bG[4];
#pragma unroll
            for (int j = 0; j < 4; ++j) {
                const int nrow = nbase + j * 16 + lrow;
                bF[j] = ldfrag_g(Wft + (size_t)nrow * DIM + ka);
                bG[j] = ldfrag_g(Wgt + (size_t)nrow * DIM + ka);
            }
#pragma unroll
            for (int j = 0; j < 4; ++j) {
                fa[j] = wmma_bf16(a, bF[j], fa[j]);
                ga[j] = wmma_bf16(a, bG[j], ga[j]);
            }
        }
    }

    // gated activation -> LDS (bf16)
#pragma unroll
    for (int j = 0; j < 4; ++j) {
        const int n = nbase + j * 16 + lrow;
        const float bfv = bf[n], bgv = bg[n];
#pragma unroll
        for (int r = 0; r < 8; ++r) {
            float fv = fa[j][r] + bfv;
            float gv = ga[j][r] + bgv;
            float u  = tanhf(fv) * (1.f / (1.f + __expf(-gv)));
            U[(mtile * 16 + lhalf * 8 + r) * ASTR + n] = f2bf(u);
        }
    }
    __syncthreads();

    // stage 2: r/s GEMM  (K = 256)
    f32x8 ra[4], sa[4];
#pragma unroll
    for (int j = 0; j < 4; ++j) { ra[j] = vzero; sa[j] = vzero; }
    for (int k0 = 0; k0 < DIM; k0 += 32) {
        const int ka = k0 + lhalf * 8;
        Frag a;
        a.q[0] = *reinterpret_cast<const uint4*>(&U[arow + ka]);
        a.q[1] = *reinterpret_cast<const uint4*>(&U[arow + ka + 16]);
        Frag bR[4], bS[4];
#pragma unroll
        for (int j = 0; j < 4; ++j) {
            const int nrow = nbase + j * 16 + lrow;
            bR[j] = ldfrag_g(Wr + (size_t)nrow * DIM + ka);
            bS[j] = ldfrag_g(Ws + (size_t)nrow * DIM + ka);
        }
#pragma unroll
        for (int j = 0; j < 4; ++j) {
            ra[j] = wmma_bf16(a, bR[j], ra[j]);
            sa[j] = wmma_bf16(a, bS[j], sa[j]);
        }
    }

    // residual + skip
#pragma unroll
    for (int j = 0; j < 4; ++j) {
        const int n = nbase + j * 16 + lrow;
        const float brv = br[n], bsv = bs[n];
#pragma unroll
        for (int r = 0; r < 8; ++r) {
            size_t idx = (size_t)(pos0 + mtile * 16 + lhalf * 8 + r) * DIM + n;
            z[idx] += ra[j][r] + brv;
            float sv = sa[j][r] + bsv;
            if (first) sum_out[idx] = sv;
            else       sum_out[idx] += sv;
        }
    }
}

// ---------------- generic bf16 WMMA GEMM (head MLP) ----------------
// out(M,N) = act(A(M,K) * W(N,K)^T + bias); block = 32 rows x (NT*64) cols
template <int NT, int ACT, int OUTBF>
__global__ __launch_bounds__(256)
void gemm_kernel(const u16* __restrict__ A, const u16* __restrict__ W,
                 const float* __restrict__ bias,
                 float* __restrict__ outF, u16* __restrict__ outB,
                 int N, int K) {
    const int tid   = threadIdx.x;
    const int wave  = tid >> 5, lane = tid & 31;
    const int mtile = wave & 1;
    const int pos0  = blockIdx.x * 32;
    const int nbase = blockIdx.y * (NT * 64) + (wave >> 1) * (NT * 16);
    const int lrow  = lane & 15, lhalf = lane >> 4;

    const f32x8 vzero = {};
    f32x8 acc[NT];
#pragma unroll
    for (int j = 0; j < NT; ++j) acc[j] = vzero;

    const u16* Ap = A + (size_t)(pos0 + mtile * 16 + lrow) * K + lhalf * 8;
    for (int k0 = 0; k0 < K; k0 += 32) {
        Frag a = ldfrag_g(Ap + k0);
        Frag b[NT];
#pragma unroll
        for (int j = 0; j < NT; ++j) {
            const int nrow = nbase + j * 16 + lrow;
            b[j] = ldfrag_g(W + (size_t)nrow * K + lhalf * 8 + k0);
        }
#pragma unroll
        for (int j = 0; j < NT; ++j)
            acc[j] = wmma_bf16(a, b[j], acc[j]);
    }
#pragma unroll
    for (int j = 0; j < NT; ++j) {
        const int n = nbase + j * 16 + lrow;
        const float bv = bias[n];
#pragma unroll
        for (int r = 0; r < 8; ++r) {
            float v = acc[j][r] + bv;
            if (ACT) v = fmaxf(v, 0.f);
            size_t idx = (size_t)(pos0 + mtile * 16 + lhalf * 8 + r) * N + n;
            if (OUTBF) outB[idx] = f2bf(v);
            else       outF[idx] = v;
        }
    }
}

// ---------------- launch ----------------
extern "C" void kernel_launch(void* const* d_in, const int* in_sizes, int n_in,
                              void* d_out, int out_size, void* d_ws, size_t ws_size,
                              hipStream_t stream) {
    (void)in_sizes; (void)n_in; (void)out_size; (void)ws_size;
    const float* x      = (const float*)d_in[0];
    const float* W_in   = (const float*)d_in[1];
    const float* b_in   = (const float*)d_in[2];
    const float* W_f    = (const float*)d_in[3];
    const float* b_f    = (const float*)d_in[4];
    const float* W_g    = (const float*)d_in[5];
    const float* b_g    = (const float*)d_in[6];
    const float* W_r    = (const float*)d_in[7];
    const float* b_r    = (const float*)d_in[8];
    const float* W_s    = (const float*)d_in[9];
    const float* b_s    = (const float*)d_in[10];
    const float* norm_w = (const float*)d_in[11];
    const float* head_w = (const float*)d_in[12];
    const float* W1     = (const float*)d_in[13];
    const float* b1     = (const float*)d_in[14];
    const float* W2     = (const float*)d_in[15];
    const float* b2     = (const float*)d_in[16];
    float* out = (float*)d_out;

    char* ws = (char*)d_ws;
    size_t off = 0;
    auto alloc = [&](size_t bytes) -> void* {
        void* p = ws + off;
        off = (off + bytes + 255) & ~(size_t)255;
        return p;
    };
    float* z   = (float*)alloc((size_t)NPOS * DIM * 4);          // 64 MiB
    float* sum = (float*)alloc((size_t)NPOS * DIM * 4);          // 64 MiB
    u16*  zn   = (u16*)  alloc((size_t)NPOS * DIM * 2);          // 32 MiB (reused as head "on")
    u16*  hbuf = (u16*)  alloc((size_t)NPOS * HIDDEN * 2);       // 128 MiB
    u16*  Wfb  = (u16*)  alloc((size_t)N_LAYERS * 2 * DIM * DIM * 2);
    u16*  Wgb  = (u16*)  alloc((size_t)N_LAYERS * 2 * DIM * DIM * 2);
    u16*  Wrb  = (u16*)  alloc((size_t)N_LAYERS * DIM * DIM * 2);
    u16*  Wsb  = (u16*)  alloc((size_t)N_LAYERS * DIM * DIM * 2);
    u16*  W1b  = (u16*)  alloc((size_t)HIDDEN * DIM * 2);
    u16*  W2b  = (u16*)  alloc((size_t)SIZE_OUT * HIDDEN * 2);

    // weight conversion (bf16, N-major / K-contiguous; taps split for f/g)
    cvt_wfg_kernel<<<N_LAYERS * 2 * DIM * DIM / 256, 256, 0, stream>>>(W_f, Wfb);
    cvt_wfg_kernel<<<N_LAYERS * 2 * DIM * DIM / 256, 256, 0, stream>>>(W_g, Wgb);
    cvt_flat_kernel<<<N_LAYERS * DIM * DIM / 256, 256, 0, stream>>>(W_r, Wrb, N_LAYERS * DIM * DIM);
    cvt_flat_kernel<<<N_LAYERS * DIM * DIM / 256, 256, 0, stream>>>(W_s, Wsb, N_LAYERS * DIM * DIM);
    cvt_flat_kernel<<<HIDDEN * DIM / 256, 256, 0, stream>>>(W1, W1b, HIDDEN * DIM);
    cvt_flat_kernel<<<SIZE_OUT * HIDDEN / 256, 256, 0, stream>>>(W2, W2b, SIZE_OUT * HIDDEN);

    input_conv_kernel<<<NPOS / 8, 256, 0, stream>>>(x, W_in, b_in, z);

    static const int DIL[N_LAYERS] = {1, 2, 4, 8, 16, 32, 64, 128, 256, 512,
                                      1, 2, 4, 8, 16, 32, 64, 128, 256, 512};
    for (int l = 0; l < N_LAYERS; ++l) {
        rmsnorm_kernel<false><<<NPOS / 8, 256, 0, stream>>>(z, norm_w + (size_t)l * DIM, zn);
        layer_kernel<<<NPOS / 32, 256, 0, stream>>>(
            zn, z, sum,
            Wfb + (size_t)l * 2 * DIM * DIM, Wgb + (size_t)l * 2 * DIM * DIM,
            Wrb + (size_t)l * DIM * DIM,     Wsb + (size_t)l * DIM * DIM,
            b_f + (size_t)l * DIM, b_g + (size_t)l * DIM,
            b_r + (size_t)l * DIM, b_s + (size_t)l * DIM,
            DIL[l], l == 0 ? 1 : 0);
    }

    // head: relu -> rmsnorm -> W1+relu -> W2
    rmsnorm_kernel<true><<<NPOS / 8, 256, 0, stream>>>(sum, head_w, zn);
    gemm_kernel<8, 1, 1><<<dim3(NPOS / 32, HIDDEN / 512), 256, 0, stream>>>(
        zn, W1b, b1, nullptr, hbuf, HIDDEN, DIM);
    gemm_kernel<2, 0, 0><<<dim3(NPOS / 32, 1), 256, 0, stream>>>(
        hbuf, W2b, b2, out, nullptr, SIZE_OUT, HIDDEN);
}
